// CapsuleNetwork_37718402793880
// MI455X (gfx1250) — matrix operand order repaired and checked by reading, not compiled, not run on hardware
//
#include <hip/hip_runtime.h>
#include <hip/hip_bf16.h>
#include <math.h>

// ---------------- CDNA5 WMMA fragment types ----------------
typedef _Float16 v16h __attribute__((ext_vector_type(16)));
typedef _Float16 v8h  __attribute__((ext_vector_type(8)));
typedef float    v8f  __attribute__((ext_vector_type(8)));

union AF { v16h v; v8h h[2]; };

__device__ __forceinline__ v8f wmma32(v16h a, v16h b, v8f c) {
    // D = A(16x32 f16) * B(32x16 f16) + C(16x16 f32)
    return __builtin_amdgcn_wmma_f32_16x16x32_f16(
        /*neg_a=*/false, a, /*neg_b=*/false, b,
        /*c_mod=*/(short)0, c, /*reuse_a=*/false, /*reuse_b=*/false);
}

// Async global->LDS copy path (CDNA5): guarded so a missing builtin
// falls back to the VGPR-staged copy without breaking the compile.
#if defined(__has_builtin)
#if __has_builtin(__builtin_amdgcn_global_load_async_to_lds_b128)
#define HAVE_ASYNC_LDS 1
#endif
#if __has_builtin(__builtin_amdgcn_s_wait_asynccnt)
#define HAVE_WAIT_ASYNC_BUILTIN 1
#endif
#endif

#ifdef HAVE_ASYNC_LDS
// Builtin signature (from hipcc diagnostic): param0 is
// "__attribute__((__vector_size__(4*sizeof(int)))) int __device__ *"
// i.e. gcc-vector int4 in AS(1); LDS side is the AS(3) counterpart.
typedef int ai4 __attribute__((vector_size(16)));
typedef __attribute__((address_space(1))) ai4* g_ai4p;
typedef __attribute__((address_space(3))) ai4* l_ai4p;
#endif

// ---------------- constants ----------------
#define BATCH   256
#define N_PRIM  1152
#define N_SEC   10
#define D_PRIM  8
#define D_SEC   16
#define EPSV    1e-7f

#define M1 (BATCH*20*20)   // 102400 rows, conv1 implicit GEMM
#define M2 (BATCH*6*6)     // 9216 rows, conv2 implicit GEMM
#define K2CH 648           // 20736 / 32

// ------------------------------------------------------------------
// Swizzle an fp32 [K][N] row-major weight matrix into per-lane WMMA
// B-operand layout: out[ck][nt][lane][16 halves]; h=lane/16, j<8 -> K=ck*32+8h+j,
// j>=8 -> K=ck*32+16+8h+(j-8); n = nt*16 + (lane&15). Zero-pads K/N.
// ------------------------------------------------------------------
__global__ void k_swizzleB(const float* __restrict__ W, _Float16* __restrict__ out,
                           int K, int N, int kchunks, int ntp) {
    int idx = blockIdx.x * blockDim.x + threadIdx.x;
    int total = kchunks * ntp * 512;
    if (idx >= total) return;
    int j  = idx & 15;
    int l  = (idx >> 4) & 31;
    int rest = idx >> 9;
    int nt = rest % ntp;
    int ck = rest / ntp;
    int k = ck * 32 + ((l >= 16) ? 8 : 0) + ((j < 8) ? j : j + 8);
    int n = nt * 16 + (l & 15);
    float v = (k < K && n < N) ? W[(size_t)k * N + n] : 0.f;
    out[idx] = (_Float16)v;
}

// ------------------------------------------------------------------
// conv1 as implicit GEMM: M=102400 (b,oy,ox), N=256, K=81 pad 96.
// grid (6400, 2), block 256 (8 waves, one 16x16 N-tile each).
// ------------------------------------------------------------------
__global__ void __launch_bounds__(256) k_conv1(const float* __restrict__ x,
        const _Float16* __restrict__ w1s, const float* __restrict__ b1,
        _Float16* __restrict__ y) {
    __shared__ _Float16 aS[16 * 32];
    const int tid  = threadIdx.x;
    const int lane = tid & 31, wv = tid >> 5;
    const int mbase = blockIdx.x * 16;
    const int ntile = blockIdx.y * 8 + wv;
    const int kb = (lane >= 16) ? 8 : 0;
    v8f acc = {};
    for (int ck = 0; ck < 3; ++ck) {
        // im2col staging: 16 rows x 32 K-taps (f32 -> f16)
        for (int e = tid; e < 512; e += 256) {
            int m = e >> 5, kk = e & 31, k = ck * 32 + kk;
            float v = 0.f;
            if (k < 81) {
                int r = mbase + m;
                int b = r / 400, rem = r % 400;
                int oy = rem / 20, ox = rem % 20;
                int ky = k / 9, kx = k % 9;
                v = x[((size_t)b * 28 + oy + ky) * 28 + ox + kx];
            }
            aS[e] = (_Float16)v;
        }
        __syncthreads();
        const _Float16* rowp = aS + (lane & 15) * 32;
        AF a;
        a.h[0] = *(const v8h*)(rowp + kb);
        a.h[1] = *(const v8h*)(rowp + kb + 16);
        v16h bf = *(const v16h*)(w1s + ((size_t)(ck * 16 + ntile) * 32 + lane) * 16);
        acc = wmma32(a.v, bf, acc);
        __syncthreads();
    }
    int n = ntile * 16 + (lane & 15);
    float bias = b1[n];
#pragma unroll
    for (int rr = 0; rr < 8; ++rr) {
        int m = mbase + ((lane < 16) ? rr : rr + 8);
        float v = acc[rr] + bias;
        v = v > 0.f ? v : 0.f;               // ReLU
        y[(size_t)m * 256 + n] = (_Float16)v;
    }
}

// ------------------------------------------------------------------
// primary-caps conv as implicit GEMM: M=9216, N=256, K=81*256.
// Block tile 32M x 128N, 8 waves; each wave keeps TWO accumulators
// (M-subtiles) sharing one B-fragment -> halves B traffic per FLOP.
// Outer loop over 81 taps; stage 32x256 f16 slab (16KB) in LDS via
// async global->LDS copies (ASYNCcnt) when available; inner 8 chunks
// x 2 WMMAs. grid (288, 2), block 256.
// ------------------------------------------------------------------
__global__ void __launch_bounds__(256) k_conv2(const _Float16* __restrict__ x1,
        const _Float16* __restrict__ w2s, const float* __restrict__ b2,
        _Float16* __restrict__ u) {
    __shared__ _Float16 aS[32 * 256];
    const int tid  = threadIdx.x;
    const int lane = tid & 31, wv = tid >> 5;
    const int mbase = blockIdx.x * 32;
    const int ntile = blockIdx.y * 8 + wv;
    const int kb = (lane >= 16) ? 8 : 0;
    // staging roles: 512 (row,seg) tasks of 32B, 2 per thread; row bases
    // are tap-invariant so precompute them.
    const _Float16* srcb[2];
    _Float16* dstb[2];
#pragma unroll
    for (int q = 0; q < 2; ++q) {
        int e = tid + q * 256;
        int ms = e >> 4, seg = e & 15;
        int r = mbase + ms;
        int b = r / 36, rem = r % 36;
        int oy = rem / 6, ox = rem % 6;
        srcb[q] = x1 + (((size_t)b * 20 + 2 * oy) * 20 + 2 * ox) * 256 + seg * 16;
        dstb[q] = aS + ms * 256 + seg * 16;
    }
    v8f acc0 = {}, acc1 = {};
    for (int tap = 0; tap < 81; ++tap) {
        int ky = tap / 9, kx = tap % 9;
        const size_t poff = (size_t)(ky * 20 + kx) * 256;
#ifdef HAVE_ASYNC_LDS
#pragma unroll
        for (int q = 0; q < 2; ++q) {
            const _Float16* s = srcb[q] + poff;
            // 32B per thread: two async B128 transfers; offset applies to
            // both global and LDS addresses per the ISA pseudocode.
            __builtin_amdgcn_global_load_async_to_lds_b128(
                (g_ai4p)s, (l_ai4p)dstb[q], 0, 0);
            __builtin_amdgcn_global_load_async_to_lds_b128(
                (g_ai4p)s, (l_ai4p)dstb[q], 16, 0);
        }
#ifdef HAVE_WAIT_ASYNC_BUILTIN
        __builtin_amdgcn_s_wait_asynccnt(0);
#else
        asm volatile("s_wait_asynccnt 0x0" ::: "memory");
#endif
#else
#pragma unroll
        for (int q = 0; q < 2; ++q) {
            const uint4* s = (const uint4*)(srcb[q] + poff);
            uint4* d = (uint4*)dstb[q];
            d[0] = s[0];
            d[1] = s[1];
        }
#endif
        __builtin_prefetch(srcb[0] + poff + 256, 0, 1);  // next pixel
        __syncthreads();
        const _Float16* row0 = aS + (lane & 15) * 256;
        const _Float16* row1 = aS + (16 + (lane & 15)) * 256;
#pragma unroll
        for (int ci = 0; ci < 8; ++ci) {
            int ck = tap * 8 + ci;
            v16h bf = *(const v16h*)(w2s + ((size_t)(ck * 16 + ntile) * 32 + lane) * 16);
            AF a0, a1;
            a0.h[0] = *(const v8h*)(row0 + ci * 32 + kb);
            a0.h[1] = *(const v8h*)(row0 + ci * 32 + kb + 16);
            a1.h[0] = *(const v8h*)(row1 + ci * 32 + kb);
            a1.h[1] = *(const v8h*)(row1 + ci * 32 + kb + 16);
            acc0 = wmma32(a0.v, bf, acc0);
            acc1 = wmma32(a1.v, bf, acc1);
        }
        __syncthreads();
    }
    int n = ntile * 16 + (lane & 15);
    float bias = b2[n];
#pragma unroll
    for (int rr = 0; rr < 8; ++rr) {
        int m0 = mbase + ((lane < 16) ? rr : rr + 8);
        u[(size_t)m0 * 256 + n] = (_Float16)(acc0[rr] + bias);
        u[(size_t)(m0 + 16) * 256 + n] = (_Float16)(acc1[rr] + bias);
    }
}

// ------------------------------------------------------------------
// Fused dynamic routing: one block per batch element. u (f16) and the
// routing logits live in LDS; u_hat is recomputed from w_pose (L2-hot)
// instead of materializing a 94MB tensor. Writes v into d_out and the
// masked capsule vector (f16) for the decoder.
// ------------------------------------------------------------------
__global__ void __launch_bounds__(256) k_routing(const _Float16* __restrict__ uh,
        const float* __restrict__ wp, const float* __restrict__ y,
        float* __restrict__ vout, _Float16* __restrict__ vmask) {
    __shared__ _Float16 uS[N_PRIM * D_PRIM];   // 18432 B
    __shared__ _Float16 lS[N_PRIM * N_SEC];    // 23040 B
    __shared__ float mxS[N_PRIM];              // 4608 B
    __shared__ float dnS[N_PRIM];              // 4608 B
    __shared__ float sS[N_SEC * D_SEC];
    __shared__ float vS[N_SEC * D_SEC];
    __shared__ float facS[N_SEC];
    const int t = threadIdx.x;
    const int b = blockIdx.x;

    const uint4* src = (const uint4*)(uh + (size_t)b * N_PRIM * D_PRIM);
    uint4* dst = (uint4*)uS;
    for (int e = t; e < N_PRIM * D_PRIM / 8; e += 256) dst[e] = src[e];
    for (int e = t; e < N_PRIM * N_SEC; e += 256) lS[e] = (_Float16)0.f;
    __syncthreads();

    for (int it = 0; it < 3; ++it) {
        // per-capsule softmax stats over j
        for (int i = t; i < N_PRIM; i += 256) {
            float mx = -1e30f;
            for (int j = 0; j < N_SEC; ++j) mx = fmaxf(mx, (float)lS[i * N_SEC + j]);
            float dn = 0.f;
            for (int j = 0; j < N_SEC; ++j) dn += __expf((float)lS[i * N_SEC + j] - mx);
            mxS[i] = mx; dnS[i] = dn;
        }
        __syncthreads();
        // s[j][k] = sum_i c[i][j] * (W_pose[i,j,k,:] . u[i,:])
        if (t < N_SEC * D_SEC) {
            int j = t >> 4, k = t & 15;
            float s = 0.f;
            for (int i = 0; i < N_PRIM; ++i) {
                float c = __expf((float)lS[i * N_SEC + j] - mxS[i]) / dnS[i];
                const float* w = wp + (((size_t)i * N_SEC + j) * D_SEC + k) * D_PRIM;
                const _Float16* uu = uS + i * D_PRIM;
                float uhk = 0.f;
#pragma unroll
                for (int mq = 0; mq < D_PRIM; ++mq) uhk += w[mq] * (float)uu[mq];
                s += c * uhk;
            }
            sS[t] = s;
        }
        __syncthreads();
        if (t < N_SEC) {
            float sn2 = 0.f;
            for (int k = 0; k < D_SEC; ++k) { float x = sS[t * 16 + k]; sn2 += x * x; }
            facS[t] = sn2 / (1.f + sn2) / (sqrtf(sn2) + EPSV);
        }
        __syncthreads();
        if (t < N_SEC * D_SEC) vS[t] = facS[t >> 4] * sS[t];
        __syncthreads();
        // agreement: b[i][j] += u_hat[i,j,:] . v[j,:]
        for (int i = t; i < N_PRIM; i += 256) {
            const _Float16* uu = uS + i * D_PRIM;
            for (int j = 0; j < N_SEC; ++j) {
                const float* w = wp + ((size_t)i * N_SEC + j) * D_SEC * D_PRIM;
                float a = 0.f;
                for (int k = 0; k < D_SEC; ++k) {
                    float uhk = 0.f;
#pragma unroll
                    for (int mq = 0; mq < D_PRIM; ++mq)
                        uhk += w[k * D_PRIM + mq] * (float)uu[mq];
                    a += uhk * vS[j * 16 + k];
                }
                lS[i * N_SEC + j] = (_Float16)((float)lS[i * N_SEC + j] + a);
            }
        }
        __syncthreads();
    }
    if (t < N_SEC * D_SEC) {
        float v = vS[t];
        vout[(size_t)b * 160 + t] = v;                                   // output v
        vmask[(size_t)b * 160 + t] = (_Float16)(y[b * N_SEC + (t >> 4)] * v);
    }
}

// ------------------------------------------------------------------
// Generic dense WMMA GEMM: C = act(A * B + bias). A: f16 [M][lda],
// B pre-swizzled. block = 4 waves, each owns one 16x16 N-tile.
// ACT: 0=none->f16, 1=relu->f16, 2=sigmoid->f32
// ------------------------------------------------------------------
template <int ACT>
__global__ void __launch_bounds__(128) k_gemm(const _Float16* __restrict__ A, int lda,
        const _Float16* __restrict__ Bs, int ntp, const float* __restrict__ bias,
        int kchunks, int ntiles, int N,
        _Float16* __restrict__ Ch, float* __restrict__ Cf, int ldc) {
    const int lane = threadIdx.x & 31, wv = threadIdx.x >> 5;
    const int ntile = blockIdx.y * 4 + wv;
    if (ntile >= ntiles) return;  // wave-uniform: safe around WMMA
    const int mbase = blockIdx.x * 16;
    const _Float16* arow = A + (size_t)(mbase + (lane & 15)) * lda;
    const int kb = (lane >= 16) ? 8 : 0;
    v8f acc = {};
    for (int ck = 0; ck < kchunks; ++ck) {
        AF a;
        a.h[0] = *(const v8h*)(arow + ck * 32 + kb);
        a.h[1] = *(const v8h*)(arow + ck * 32 + kb + 16);
        v16h bf = *(const v16h*)(Bs + ((size_t)(ck * ntp + ntile) * 32 + lane) * 16);
        acc = wmma32(a.v, bf, acc);
    }
    int n = ntile * 16 + (lane & 15);
    if (n >= N) return;
    float bv = bias ? bias[n] : 0.f;
#pragma unroll
    for (int rr = 0; rr < 8; ++rr) {
        int m = mbase + ((lane < 16) ? rr : rr + 8);
        float v = acc[rr] + bv;
        if (ACT == 1) v = v > 0.f ? v : 0.f;
        if (ACT == 2) {
            v = 1.f / (1.f + __expf(-v));
            Cf[(size_t)m * ldc + n] = v;
        } else {
            Ch[(size_t)m * ldc + n] = (_Float16)v;
        }
    }
}

// ------------------------------------------------------------------
extern "C" void kernel_launch(void* const* d_in, const int* in_sizes, int n_in,
                              void* d_out, int out_size, void* d_ws, size_t ws_size,
                              hipStream_t stream) {
    const float* input_x = (const float*)d_in[0];
    const float* y       = (const float*)d_in[1];
    const float* conv1_w = (const float*)d_in[2];
    const float* conv1_b = (const float*)d_in[3];
    const float* pc_w    = (const float*)d_in[4];
    const float* pc_b    = (const float*)d_in[5];
    const float* w_pose  = (const float*)d_in[6];
    const float* d1_w    = (const float*)d_in[7];
    const float* d1_b    = (const float*)d_in[8];
    const float* d2_w    = (const float*)d_in[9];
    const float* d2_b    = (const float*)d_in[10];
    const float* d3_w    = (const float*)d_in[11];
    const float* d3_b    = (const float*)d_in[12];
    float* out = (float*)d_out;

    char* wsb = (char*)d_ws;
    size_t off = 0;
    auto alloc = [&](size_t halves) -> _Float16* {
        _Float16* p = (_Float16*)(wsb + off);
        off = (off + halves * 2 + 255) & ~(size_t)255;
        return p;
    };
    _Float16* w1s   = alloc((size_t)3 * 16 * 512);         // conv1 weights swizzled
    _Float16* w2s   = alloc((size_t)K2CH * 16 * 512);      // pc weights swizzled
    _Float16* d1s   = alloc((size_t)5 * 32 * 512);
    _Float16* d2s   = alloc((size_t)16 * 64 * 512);
    _Float16* d3s   = alloc((size_t)32 * 52 * 512);
    _Float16* x1h   = alloc((size_t)M1 * 256);             // conv1 output f16
    _Float16* uhB   = alloc((size_t)M2 * 256);             // primary caps f16
    _Float16* vmask = alloc((size_t)BATCH * 160);
    _Float16* h1    = alloc((size_t)BATCH * 512);
    _Float16* h2    = alloc((size_t)BATCH * 1024);

    auto swz = [&](const float* W, _Float16* o, int K, int N, int kc, int ntp) {
        int total = kc * ntp * 512;
        k_swizzleB<<<(total + 255) / 256, 256, 0, stream>>>(W, o, K, N, kc, ntp);
    };
    swz(conv1_w, w1s, 81, 256, 3, 16);
    swz(pc_w,    w2s, 20736, 256, K2CH, 16);
    swz(d1_w,    d1s, 160, 512, 5, 32);
    swz(d2_w,    d2s, 512, 1024, 16, 64);
    swz(d3_w,    d3s, 1024, 784, 32, 52);

    k_conv1<<<dim3(M1 / 16, 2), 256, 0, stream>>>(input_x, w1s, conv1_b, x1h);
    k_conv2<<<dim3(M2 / 32, 2), 256, 0, stream>>>(x1h, w2s, pc_b, uhB);
    k_routing<<<BATCH, 256, 0, stream>>>(uhB, w_pose, y, out, vmask);

    k_gemm<1><<<dim3(BATCH / 16, 8),  128, 0, stream>>>(vmask, 160,  d1s, 32, d1_b, 5,  32, 512,  h1, nullptr, 512);
    k_gemm<1><<<dim3(BATCH / 16, 16), 128, 0, stream>>>(h1,    512,  d2s, 64, d2_b, 16, 64, 1024, h2, nullptr, 1024);
    k_gemm<2><<<dim3(BATCH / 16, 13), 128, 0, stream>>>(h2,    1024, d3s, 52, d3_b, 32, 49, 784,  nullptr, out + (size_t)BATCH * 160, 784);
}
